// MultiHeadAttention_38354057953499
// MI455X (gfx1250) — matrix operand order repaired and checked by reading, not compiled, not run on hardware
//
#include <hip/hip_runtime.h>
#include <hip/hip_bf16.h>

// ---------------------------------------------------------------------------
// MHA forward on gfx1250 (MI455X): bf16 WMMA everywhere, flash-attention so
// the 536MB score tensor never exists. Workspace: Q,K (bf16 [BH,S,HD]),
// V^T (bf16 [BH,HD,S]), ctx (bf16 [B,S,D]) = 32MB total, L2-resident.
// GEMM waves own 32x64 tiles; k-loop is software-pipelined with double-
// buffered A/B fragments so the WMMA->VALU WAR hazard (4 coexec NOPs for
// bf16 WMMA) never triggers and cvt VALU co-executes with the matrix pipe.
// Flash waves process 64 keys per softmax pass (16 WMMAs per iteration)
// and prefetch the next K/V tiles (global_prefetch) during the softmax.
// ---------------------------------------------------------------------------

typedef __bf16 bf16;
typedef __attribute__((ext_vector_type(16))) __bf16 v16bf;
typedef __attribute__((ext_vector_type(8)))  __bf16 v8bf;
typedef __attribute__((ext_vector_type(8)))  float  v8f;
typedef __attribute__((ext_vector_type(4)))  float  v4f;

#define B_  2
#define S_  2048
#define D_  1024
#define H_  16
#define HD_ 64
#define M_  (B_ * S_)          // 4096 rows in all projection GEMMs
#define NEG_BIG (-3.0e38f)

// ---- WMMA wrapper: D = A(16x32 bf16) * B(32x16 bf16) + C(16x16 f32) -------
static __device__ __forceinline__ v8f wmma_bf16(v16bf a, v16bf b, v8f c) {
  return __builtin_amdgcn_wmma_f32_16x16x32_bf16(
      /*neg_a=*/false, a, /*neg_b=*/false, b,
      /*c_mod=*/(short)0, c, /*reuse_a=*/false, /*reuse_b=*/false);
}

// ---- A fragment (16x32): lane holds row M=L&15.
//      elements 0..7  -> K = k0 + half*8 + e
//      elements 8..15 -> K = k0 + 16 + half*8 + (e-8)
static __device__ __forceinline__ v16bf a_frag_f32(const float* __restrict__ row,
                                                   int k0, int hf) {
  v4f x0 = *(const v4f*)(row + k0 + hf * 8);
  v4f x1 = *(const v4f*)(row + k0 + hf * 8 + 4);
  v4f x2 = *(const v4f*)(row + k0 + 16 + hf * 8);
  v4f x3 = *(const v4f*)(row + k0 + 16 + hf * 8 + 4);
  v16bf a;
#pragma unroll
  for (int i = 0; i < 4; ++i) {
    a[i]      = (bf16)x0[i];
    a[4 + i]  = (bf16)x1[i];
    a[8 + i]  = (bf16)x2[i];
    a[12 + i] = (bf16)x3[i];
  }
  return a;
}

static __device__ __forceinline__ v16bf a_frag_bf16(const bf16* row, int k0, int hf) {
  v8bf lo = *(const v8bf*)(row + k0 + hf * 8);
  v8bf hi = *(const v8bf*)(row + k0 + 16 + hf * 8);
  v16bf a;
#pragma unroll
  for (int i = 0; i < 8; ++i) { a[i] = lo[i]; a[8 + i] = hi[i]; }
  return a;
}

// ---- B fragment (32x16): lane holds column N=L&15 (row pointer passed in).
//      elements 0..7  -> K = k0 + half*8 + e
//      elements 8..15 -> K = k0 + 16 + half*8 + (e-8)
static __device__ __forceinline__ v16bf b_frag_f32(const float* __restrict__ row,
                                                   int k0, int hf) {
  v4f x0 = *(const v4f*)(row + k0 + hf * 8);
  v4f x1 = *(const v4f*)(row + k0 + hf * 8 + 4);
  v4f x2 = *(const v4f*)(row + k0 + 16 + hf * 8);
  v4f x3 = *(const v4f*)(row + k0 + 16 + hf * 8 + 4);
  v16bf b;
#pragma unroll
  for (int i = 0; i < 4; ++i) {
    b[i]      = (bf16)x0[i];
    b[4 + i]  = (bf16)x1[i];
    b[8 + i]  = (bf16)x2[i];
    b[12 + i] = (bf16)x3[i];
  }
  return b;
}

static __device__ __forceinline__ v16bf b_frag_bf16(const bf16* row, int k0, int hf) {
  v8bf lo = *(const v8bf*)(row + k0 + hf * 8);
  v8bf hi = *(const v8bf*)(row + k0 + 16 + hf * 8);
  v16bf b;
#pragma unroll
  for (int i = 0; i < 8; ++i) { b[i] = lo[i]; b[8 + i] = hi[i]; }
  return b;
}

// ===========================================================================
// Kernel 1: fused QKV projection.  out[m, n] = sum_k x[m,k] * W[n,k]
// N dimension is 3*D (Wq|Wk|Wv concatenated). Q gets the 1/sqrt(HD) scale
// folded in. Q,K stored [BH,S,HD]; V stored transposed [BH,HD,S].
// Wave tile: 32(M) x 64(N). Block: 8 waves stacked in M -> 256 x 64.
// k-loop software-pipelined: fragments for k+32 load into a disjoint
// register set while the k-step WMMAs execute.
// ===========================================================================
__global__ __launch_bounds__(256) void qkv_proj_kernel(
    const float* __restrict__ x, const float* __restrict__ Wq,
    const float* __restrict__ Wk, const float* __restrict__ Wv,
    bf16* __restrict__ qw, bf16* __restrict__ kw, bf16* __restrict__ vT) {
  const int lane = threadIdx.x & 31;
  const int wave = threadIdx.x >> 5;
  const int l15  = lane & 15;
  const int hf   = lane >> 4;

  const int m0 = blockIdx.x * 256 + wave * 32;
  const int n0 = blockIdx.y * 64;
  const int proj = n0 >> 10;          // 0=Q 1=K 2=V (64 | 1024, never straddles)
  const int np   = n0 & 1023;

  const float* W = (proj == 0) ? Wq : ((proj == 1) ? Wk : Wv);
  const float* arow0 = x + (size_t)(m0 + l15) * D_;
  const float* arow1 = x + (size_t)(m0 + 16 + l15) * D_;
  const float* brow[4];
#pragma unroll
  for (int t = 0; t < 4; ++t)
    brow[t] = W + (size_t)(np + t * 16 + l15) * D_;

  v8f acc[2][4];
#pragma unroll
  for (int s = 0; s < 2; ++s)
#pragma unroll
    for (int t = 0; t < 4; ++t)
#pragma unroll
      for (int i = 0; i < 8; ++i) acc[s][t][i] = 0.0f;

  // ---- software pipeline: prologue loads k=0 fragments ----
  v16bf a0 = a_frag_f32(arow0, 0, hf);
  v16bf a1 = a_frag_f32(arow1, 0, hf);
  v16bf bc[4];
#pragma unroll
  for (int t = 0; t < 4; ++t) bc[t] = b_frag_f32(brow[t], 0, hf);

  for (int k0 = 0; k0 < D_; k0 += 32) {
    // next k-step (wraps to 0 on the last iteration: harmless dummy load)
    const int kn = (k0 + 32 < D_) ? (k0 + 32) : 0;
    v16bf na0 = a_frag_f32(arow0, kn, hf);
    v16bf na1 = a_frag_f32(arow1, kn, hf);
    v16bf nb[4];
#pragma unroll
    for (int t = 0; t < 4; ++t) nb[t] = b_frag_f32(brow[t], kn, hf);

#pragma unroll
    for (int t = 0; t < 4; ++t) {
      acc[0][t] = wmma_bf16(a0, bc[t], acc[0][t]);
      acc[1][t] = wmma_bf16(a1, bc[t], acc[1][t]);
    }
    a0 = na0;
    a1 = na1;
#pragma unroll
    for (int t = 0; t < 4; ++t) bc[t] = nb[t];
  }

  const float scale = (proj == 0) ? 0.125f : 1.0f;   // 1/sqrt(64) folded into Q
#pragma unroll
  for (int sub = 0; sub < 2; ++sub) {
#pragma unroll
    for (int t = 0; t < 4; ++t) {
      const int n  = np + t * 16 + l15;
      const int h  = n >> 6;
      const int hd = n & 63;
#pragma unroll
      for (int r = 0; r < 8; ++r) {
        const int m  = m0 + sub * 16 + r + hf * 8;
        const int bi = m >> 11;          // / S_
        const int s  = m & (S_ - 1);
        const int bh = bi * H_ + h;
        const bf16 val = (bf16)(acc[sub][t][r] * scale);
        if (proj == 2)      vT[((size_t)bh * HD_ + hd) * S_ + s] = val;
        else if (proj == 1) kw[((size_t)bh * S_ + s) * HD_ + hd] = val;
        else                qw[((size_t)bh * S_ + s) * HD_ + hd] = val;
      }
    }
  }
}

// ===========================================================================
// Kernel 2: causal flash attention. One wave owns a 16-query tile of one
// (b,h): streams keys in tiles of 64; per step 8 WMMAs for S=Q*K^T and
// 8 WMMAs for O+=P*V; online softmax with 16-lane shfl reductions; the
// score->A-operand relayout goes through a 2KB per-wave LDS bounce.
// Next key tile is prefetched (global_prefetch) during the softmax pass.
// ===========================================================================
__global__ __launch_bounds__(256) void flash_attn_kernel(
    const bf16* __restrict__ qw, const bf16* __restrict__ kw,
    const bf16* __restrict__ vT, bf16* __restrict__ ctx) {
  __shared__ bf16 ldsP[8][16 * 64];   // per-wave P tile, row-major 16x64

  const int lane = threadIdx.x & 31;
  const int wave = threadIdx.x >> 5;
  const int l15  = lane & 15;
  const int hf   = lane >> 4;

  const int g  = blockIdx.x * 8 + wave;   // global wave id
  const int bh = g >> 7;                  // / (S_/16) = /128
  const int q0 = (g & 127) * 16;

  // Q fragments for this 16x64 tile (K-dim 64 -> two A fragments), loaded once.
  const bf16* qrow = qw + ((size_t)bh * S_ + q0 + l15) * HD_;
  const v16bf aq0 = a_frag_bf16(qrow, 0, hf);
  const v16bf aq1 = a_frag_bf16(qrow, 32, hf);

  float mrun[8], lrun[8];
  v8f accO[4];
#pragma unroll
  for (int r = 0; r < 8; ++r) { mrun[r] = NEG_BIG; lrun[r] = 0.0f; }
#pragma unroll
  for (int t = 0; t < 4; ++t)
#pragma unroll
    for (int i = 0; i < 8; ++i) accO[t][i] = 0.0f;

  const int nsteps = (q0 + 79) >> 6;      // 64-key tiles covering keys 0..q0+15
  for (int st = 0; st < nsteps; ++st) {
    const int kb = st * 64;

    // ---- scores: S = Q(16x64) @ K^T(64x64), four 16-wide N tiles ----
    v8f sc[4];
#pragma unroll
    for (int t = 0; t < 4; ++t) {
#pragma unroll
      for (int i = 0; i < 8; ++i) sc[t][i] = 0.0f;
      const bf16* krow = kw + ((size_t)bh * S_ + kb + t * 16 + l15) * HD_;
      v16bf b0 = b_frag_bf16(krow, 0, hf);
      sc[t] = wmma_bf16(aq0, b0, sc[t]);
      v16bf b1 = b_frag_bf16(krow, 32, hf);
      sc[t] = wmma_bf16(aq1, b1, sc[t]);
    }

    // ---- prefetch next key tile of K and V while we do the softmax ----
    if (st + 1 < nsteps) {
      const int kn = kb + 64;
      __builtin_prefetch(kw + ((size_t)bh * S_ + kn + lane * 2) * HD_, 0, 1);
      __builtin_prefetch(vT + ((size_t)bh * HD_ + lane * 2) * S_ + kn, 0, 1);
    }

    // ---- causal mask + rowwise max over 64 keys ----
    float rmax[8];
#pragma unroll
    for (int r = 0; r < 8; ++r) {
      const int qidx = q0 + r + hf * 8;
#pragma unroll
      for (int t = 0; t < 4; ++t) {
        const int key = kb + t * 16 + l15;
        if (key > qidx) sc[t][r] = NEG_BIG;
      }
      float v = fmaxf(fmaxf(sc[0][r], sc[1][r]), fmaxf(sc[2][r], sc[3][r]));
      v = fmaxf(v, __shfl_xor(v, 1));
      v = fmaxf(v, __shfl_xor(v, 2));
      v = fmaxf(v, __shfl_xor(v, 4));
      v = fmaxf(v, __shfl_xor(v, 8));
      rmax[r] = v;
    }

    // ---- online softmax update ----
#pragma unroll
    for (int r = 0; r < 8; ++r) {
      const float mnew = fmaxf(mrun[r], rmax[r]);
      const float corr = __expf(mrun[r] - mnew);
      mrun[r] = mnew;
      lrun[r] *= corr;
#pragma unroll
      for (int t = 0; t < 4; ++t) accO[t][r] *= corr;
      float ls = 0.0f;
#pragma unroll
      for (int t = 0; t < 4; ++t) {
        const float p = __expf(sc[t][r] - mnew);
        sc[t][r] = p;
        ls += p;
      }
      ls += __shfl_xor(ls, 1);
      ls += __shfl_xor(ls, 2);
      ls += __shfl_xor(ls, 4);
      ls += __shfl_xor(ls, 8);
      lrun[r] += ls;
    }

    // ---- relayout P (C/D frag) -> A frag via per-wave LDS (same-wave DS
    //      ops are in-order; no barrier required) ----
    bf16* pl = ldsP[wave];
#pragma unroll
    for (int r = 0; r < 8; ++r) {
      const int row = r + hf * 8;
#pragma unroll
      for (int t = 0; t < 4; ++t)
        pl[row * 64 + t * 16 + l15] = (bf16)sc[t][r];
    }
    const v16bf ap0 = a_frag_bf16(pl + l15 * 64, 0, hf);
    const v16bf ap1 = a_frag_bf16(pl + l15 * 64, 32, hf);

    // ---- O += P(16x64) @ V(64x64); V^T layout makes B loads contiguous ----
#pragma unroll
    for (int t = 0; t < 4; ++t) {
      const bf16* vrow = vT + ((size_t)bh * HD_ + t * 16 + l15) * S_ + kb;
      v16bf bv0 = b_frag_bf16(vrow, 0, hf);
      accO[t] = wmma_bf16(ap0, bv0, accO[t]);
      v16bf bv1 = b_frag_bf16(vrow, 32, hf);
      accO[t] = wmma_bf16(ap1, bv1, accO[t]);
    }
  }

  // ---- epilogue: normalize and write ctx as bf16 [B,S,D] ----
  const int bi = bh >> 4;      // / H_
  const int h  = bh & 15;
#pragma unroll
  for (int t = 0; t < 4; ++t) {
#pragma unroll
    for (int r = 0; r < 8; ++r) {
      const int m = q0 + r + hf * 8;
      const float o = accO[t][r] / lrun[r];
      ctx[((size_t)bi * S_ + m) * D_ + h * HD_ + t * 16 + l15] = (bf16)o;
    }
  }
}

// ===========================================================================
// Kernel 3: output projection. y[m,n] = sum_k ctx[m,k] * Wo[n,k] + b[n]
// Wave tile: 32(M) x 64(N), software-pipelined like kernel 1.
// ===========================================================================
__global__ __launch_bounds__(256) void out_proj_kernel(
    const bf16* __restrict__ ctx, const float* __restrict__ Wo,
    const float* __restrict__ bias, float* __restrict__ y) {
  const int lane = threadIdx.x & 31;
  const int wave = threadIdx.x >> 5;
  const int l15  = lane & 15;
  const int hf   = lane >> 4;

  const int m0 = blockIdx.x * 256 + wave * 32;
  const int n0 = blockIdx.y * 64;

  const bf16* arow0 = ctx + (size_t)(m0 + l15) * D_;
  const bf16* arow1 = ctx + (size_t)(m0 + 16 + l15) * D_;
  const float* brow[4];
#pragma unroll
  for (int t = 0; t < 4; ++t)
    brow[t] = Wo + (size_t)(n0 + t * 16 + l15) * D_;

  v8f acc[2][4];
#pragma unroll
  for (int s = 0; s < 2; ++s)
#pragma unroll
    for (int t = 0; t < 4; ++t)
#pragma unroll
      for (int i = 0; i < 8; ++i) acc[s][t][i] = 0.0f;

  // ---- software pipeline: prologue loads k=0 fragments ----
  v16bf a0 = a_frag_bf16(arow0, 0, hf);
  v16bf a1 = a_frag_bf16(arow1, 0, hf);
  v16bf bc[4];
#pragma unroll
  for (int t = 0; t < 4; ++t) bc[t] = b_frag_f32(brow[t], 0, hf);

  for (int k0 = 0; k0 < D_; k0 += 32) {
    const int kn = (k0 + 32 < D_) ? (k0 + 32) : 0;
    v16bf na0 = a_frag_bf16(arow0, kn, hf);
    v16bf na1 = a_frag_bf16(arow1, kn, hf);
    v16bf nb[4];
#pragma unroll
    for (int t = 0; t < 4; ++t) nb[t] = b_frag_f32(brow[t], kn, hf);

#pragma unroll
    for (int t = 0; t < 4; ++t) {
      acc[0][t] = wmma_bf16(a0, bc[t], acc[0][t]);
      acc[1][t] = wmma_bf16(a1, bc[t], acc[1][t]);
    }
    a0 = na0;
    a1 = na1;
#pragma unroll
    for (int t = 0; t < 4; ++t) bc[t] = nb[t];
  }

#pragma unroll
  for (int sub = 0; sub < 2; ++sub) {
#pragma unroll
    for (int t = 0; t < 4; ++t) {
      const int n = n0 + t * 16 + l15;
      const float bs = bias[n];
#pragma unroll
      for (int r = 0; r < 8; ++r) {
        const int m = m0 + sub * 16 + r + hf * 8;
        y[(size_t)m * D_ + n] = acc[sub][t][r] + bs;
      }
    }
  }
}

// ===========================================================================
extern "C" void kernel_launch(void* const* d_in, const int* in_sizes, int n_in,
                              void* d_out, int out_size, void* d_ws, size_t ws_size,
                              hipStream_t stream) {
  (void)in_sizes; (void)n_in; (void)out_size; (void)ws_size;
  const float* x  = (const float*)d_in[0];
  const float* Wq = (const float*)d_in[1];
  const float* Wk = (const float*)d_in[2];
  const float* Wv = (const float*)d_in[3];
  const float* Wo = (const float*)d_in[4];
  const float* bo = (const float*)d_in[5];
  float* y = (float*)d_out;

  // Workspace layout (bf16): Q | K | V^T | ctx  -> 4 * 8MB = 32MB
  const size_t elems = (size_t)M_ * D_;
  char* ws = (char*)d_ws;
  bf16* qw = (bf16*)(ws);
  bf16* kw = (bf16*)(ws + 2 * elems);
  bf16* vT = (bf16*)(ws + 4 * elems);
  bf16* cx = (bf16*)(ws + 6 * elems);

  // 1) fused QKV projection: grid (M/256, 3N/64) = (16, 48)
  qkv_proj_kernel<<<dim3(M_ / 256, (3 * D_) / 64), 256, 0, stream>>>(
      x, Wq, Wk, Wv, qw, kw, vT);

  // 2) flash attention: 32 (b,h) * 128 q-tiles = 4096 waves / 8 per block
  flash_attn_kernel<<<dim3((B_ * H_ * (S_ / 16)) / 8), 256, 0, stream>>>(
      qw, kw, vT, cx);

  // 3) output projection: grid (16, 16)
  out_proj_kernel<<<dim3(M_ / 256, D_ / 64), 256, 0, stream>>>(cx, Wo, bo, y);
}
